// PBFAFeatureMixAttention_60103772340309
// MI455X (gfx1250) — matrix operand order repaired and checked by reading, not compiled, not run on hardware
//
#include <hip/hip_runtime.h>

// PBFA feature-mix attention for MI455X (gfx1250, wave32, WMMA).
// B=4, S=8192, D=1024, H=16, Dh=64, MAX_N=6, F=13.
// Heavy GEMMs: bf16 WMMA (v_wmma_f32_16x16x32_bf16) with f32 accum,
// triple-buffered GLOBAL_LOAD_ASYNC_TO_LDS_B128 pipeline (ASYNCcnt).

#define BB   4
#define SS   8192
#define DD   1024
#define HH   16
#define DHH  64
#define FF   13
#define MTOT (BB * SS)        // 32768 rows
#define NQKV (3 * DD)         // 3072

typedef __attribute__((ext_vector_type(16))) __bf16 v16bf;
typedef __attribute__((ext_vector_type(8)))  float  v8f;

union BF16x16 { uint4 u[2]; v16bf v; };
union F32x8   { v8f v; float f[8]; };

// One 16-byte async copy global -> LDS, tracked by ASYNCcnt.
__device__ __forceinline__ void async_copy16(uint32_t ldsOff, const void* gaddr)
{
    asm volatile("global_load_async_to_lds_b128 %0, %1, off"
                 :: "v"(ldsOff), "v"(gaddr) : "memory");
}

// ---------------------------------------------------------------------------
// Tiled bf16 WMMA GEMM:  C(M x N) = A(M x K, row-major bf16) * Bt(N x K)^T
// Block tile 128x256, BK=32, 256 threads = 8 waves (2 M x 4 N),
// wave tile 64x64 = 4x4 WMMA 16x16 accumulators (16 WMMA / K-step / wave).
// Triple-buffered async global->LDS pipeline: issue tile it+2 after the
// barrier of iteration it; s_wait_asynccnt 6 guarantees tile it is resident
// (async loads complete in order; 6 asyncs issued per tile per wave).
// EPI==0: QKV epilogue (bias, per-head L2 norm for q/k, bf16 (B,H,S,Dh))
// EPI==1: output epilogue (bias add, f32 store to d_out)
// ---------------------------------------------------------------------------
#define BUFBYTES 24576   // 128x32 A (8KB) + 256x32 B (16KB) bf16
#define CT_PITCH 257     // 257 % 64 == 1 -> conflict-free column sweeps

template <int EPI>
__global__ __launch_bounds__(256) void wmma_gemm_kernel(
    const __bf16* __restrict__ A, const __bf16* __restrict__ Bt,
    int K, int N, const float* __restrict__ bias,
    __bf16* __restrict__ qn, __bf16* __restrict__ kn, __bf16* __restrict__ vb,
    float* __restrict__ outp)
{
    extern __shared__ char smem[];
    float* Ct = (float*)smem;                      // 128 x 257 f32 (reused)

    const int t    = threadIdx.x;
    const int wave = t >> 5;
    const int lane = t & 31;
    const int lrow = lane & 15;
    const int hi   = lane >> 4;                    // lane half: 0 or 1
    const int gm0  = blockIdx.y * 128;
    const int gn0  = blockIdx.x * 256;
    const int wm   = (wave >> 2) * 64;             // wave M offset in tile
    const int wn   = (wave & 3) * 64;              // wave N offset in tile
    const uint32_t smem32 = (uint32_t)(uintptr_t)smem;  // AS(3) byte offset

    // Per-thread copy slots (A: 2 x 16B, B: 4 x 16B per K-tile).
    const int aslot_r0 = (t + 0 * 256) >> 2, aslot_c0 = ((t + 0 * 256) & 3) * 8;
    const int aslot_r1 = (t + 1 * 256) >> 2, aslot_c1 = ((t + 1 * 256) & 3) * 8;

    auto issue_tile = [&](int kt) {
        const int kk = kt * 32;
        const uint32_t base = smem32 + (uint32_t)(kt % 3) * BUFBYTES;
        async_copy16(base + (uint32_t)(aslot_r0 * 32 + aslot_c0) * 2,
                     &A[(size_t)(gm0 + aslot_r0) * K + kk + aslot_c0]);
        async_copy16(base + (uint32_t)(aslot_r1 * 32 + aslot_c1) * 2,
                     &A[(size_t)(gm0 + aslot_r1) * K + kk + aslot_c1]);
#pragma unroll
        for (int i = 0; i < 4; ++i) {
            int slot = t + i * 256;
            int r = slot >> 2, c = (slot & 3) * 8;
            async_copy16(base + 8192u + (uint32_t)(r * 32 + c) * 2,
                         &Bt[(size_t)(gn0 + r) * K + kk + c]);
        }
    };

    F32x8 acc[4][4];
#pragma unroll
    for (int i = 0; i < 4; ++i)
#pragma unroll
        for (int j = 0; j < 4; ++j)
#pragma unroll
            for (int r = 0; r < 8; ++r) acc[i][j].f[r] = 0.0f;

    const int NK = K / 32;
    issue_tile(0);
    issue_tile(1);

    for (int it = 0; it < NK; ++it) {
        // Wave-local: tile `it` resident once <=6 asyncs (tile it+1) remain.
        asm volatile("s_wait_asynccnt 0x6" ::: "memory");
        __syncthreads();   // all waves' tile-it loads done; prev compute done
        if (it + 2 < NK) issue_tile(it + 2);

        const __bf16* As = (const __bf16*)(smem + (it % 3) * BUFBYTES);
        const __bf16* Bs = As + 4096;

        // B fragments: 32x16 bf16, lane holds column n=lrow, K = hi*16 + 0..15
        BF16x16 bfrag[4];
#pragma unroll
        for (int nt = 0; nt < 4; ++nt) {
            const __bf16* bp = &Bs[(wn + nt * 16 + lrow) * 32 + hi * 16];
            bfrag[nt].u[0] = *(const uint4*)bp;
            bfrag[nt].u[1] = *(const uint4*)(bp + 8);
        }
        // A fragments: 16x32 bf16, lane row m=lrow, K = hi*16 + 0..15
#pragma unroll
        for (int mt = 0; mt < 4; ++mt) {
            BF16x16 afrag;
            const __bf16* ap = &As[(wm + mt * 16 + lrow) * 32];
            afrag.u[0] = *(const uint4*)(ap + hi * 16);
            afrag.u[1] = *(const uint4*)(ap + hi * 16 + 8);
#pragma unroll
            for (int nt = 0; nt < 4; ++nt) {
                acc[mt][nt].v = __builtin_amdgcn_wmma_f32_16x16x32_bf16(
                    false, afrag.v, false, bfrag[nt].v,
                    (short)0, acc[mt][nt].v, false, false);
            }
        }
    }
    __syncthreads();   // everyone done reading LDS buffers -> reuse as Ct

    // Stage accumulators (D layout: m = r + hi*8, n = lrow), pitch 257.
#pragma unroll
    for (int mt = 0; mt < 4; ++mt)
#pragma unroll
        for (int nt = 0; nt < 4; ++nt)
#pragma unroll
            for (int r = 0; r < 8; ++r)
                Ct[(wm + mt * 16 + hi * 8 + r) * CT_PITCH + wn + nt * 16 + lrow] =
                    acc[mt][nt].f[r];
    __syncthreads();

    // Epilogue: 512 (row, 64-col group) tasks, 2 per thread.
#pragma unroll
    for (int task = t; task < 512; task += 256) {
        const int row   = task >> 2;
        const int g     = task & 3;
        const int gr    = gm0 + row;
        const int gcol0 = gn0 + g * 64;
        const float* crow = &Ct[row * CT_PITCH + g * 64];

        if (EPI == 0) {
            const int mat  = gcol0 >> 10;            // 0=q, 1=k, 2=v
            const int head = (gcol0 & 1023) >> 6;
            const int b    = gr >> 13;               // gr / S
            const int s    = gr & (SS - 1);
            const size_t dstoff = ((size_t)(b * HH + head) * SS + s) * DHH;
            if (mat < 2) {
                float ssum = 0.0f;
                for (int j = 0; j < 64; ++j) {
                    float v = crow[j] + bias[gcol0 + j];
                    ssum += v * v;
                }
                const float inv = 1.0f / fmaxf(sqrtf(ssum), 1e-12f);
                __bf16* dst = (mat == 0 ? qn : kn) + dstoff;
#pragma unroll
                for (int c8 = 0; c8 < 8; ++c8) {
                    union { uint4 u; __bf16 h[8]; } pk;
#pragma unroll
                    for (int j = 0; j < 8; ++j)
                        pk.h[j] = (__bf16)((crow[c8 * 8 + j] + bias[gcol0 + c8 * 8 + j]) * inv);
                    *(uint4*)&dst[c8 * 8] = pk.u;
                }
            } else {
                __bf16* dst = vb + dstoff;
#pragma unroll
                for (int c8 = 0; c8 < 8; ++c8) {
                    union { uint4 u; __bf16 h[8]; } pk;
#pragma unroll
                    for (int j = 0; j < 8; ++j)
                        pk.h[j] = (__bf16)(crow[c8 * 8 + j] + bias[gcol0 + c8 * 8 + j]);
                    *(uint4*)&dst[c8 * 8] = pk.u;
                }
            }
        } else {
            float* dst = outp + (size_t)gr * N + gcol0;
#pragma unroll
            for (int c4 = 0; c4 < 16; ++c4) {
                float4 v;
                v.x = crow[c4 * 4 + 0] + bias[gcol0 + c4 * 4 + 0];
                v.y = crow[c4 * 4 + 1] + bias[gcol0 + c4 * 4 + 1];
                v.z = crow[c4 * 4 + 2] + bias[gcol0 + c4 * 4 + 2];
                v.w = crow[c4 * 4 + 3] + bias[gcol0 + c4 * 4 + 3];
                *(float4*)&dst[c4 * 4] = v;
            }
        }
    }
}

// ---------------------------------------------------------------------------
// f32 -> bf16 bulk convert (4 elems/thread)
// ---------------------------------------------------------------------------
__global__ __launch_bounds__(256) void cvt_x_kernel(
    const float* __restrict__ src, __bf16* __restrict__ dst, int n4)
{
    int i = blockIdx.x * blockDim.x + threadIdx.x;
    if (i < n4) {
        float4 v = ((const float4*)src)[i];
        union { uint2 u; __bf16 h[4]; } pk;
        pk.h[0] = (__bf16)v.x; pk.h[1] = (__bf16)v.y;
        pk.h[2] = (__bf16)v.z; pk.h[3] = (__bf16)v.w;
        ((uint2*)dst)[i] = pk.u;
    }
}

// ---------------------------------------------------------------------------
// Transpose + convert: src f32 (K x N row-major)  ->  dst bf16 (N x K)
// ---------------------------------------------------------------------------
__global__ void transpose_cvt_kernel(
    const float* __restrict__ src, __bf16* __restrict__ dst, int K, int N)
{
    __shared__ float tile[32][33];
    const int bx = blockIdx.x * 32, by = blockIdx.y * 32;
    const int tx = threadIdx.x, ty = threadIdx.y;   // block (32, 8)
    for (int i = 0; i < 32; i += 8)
        tile[ty + i][tx] = src[(size_t)(by + ty + i) * N + bx + tx];
    __syncthreads();
    for (int i = 0; i < 32; i += 8)
        dst[(size_t)(bx + ty + i) * K + by + tx] = (__bf16)tile[tx][ty + i];
}

// ---------------------------------------------------------------------------
// Build concatenated qkv bias (3072)
// ---------------------------------------------------------------------------
__global__ void bias_kernel(const float* __restrict__ bq, const float* __restrict__ bk,
                            const float* __restrict__ bv, float* __restrict__ outb)
{
    int i = blockIdx.x * blockDim.x + threadIdx.x;
    if (i < NQKV)
        outb[i] = i < DD ? bq[i] : (i < 2 * DD ? bk[i - DD] : bv[i - 2 * DD]);
}

// ---------------------------------------------------------------------------
// Per-head sqrt(softmax(degree_logits)) expanded to 13 feature slots
// ---------------------------------------------------------------------------
__global__ void fw_kernel(const float* __restrict__ logits, float* __restrict__ fw)
{
    int h = threadIdx.x;
    if (h < HH) {
        float l[7];
        float m = -1e30f;
        for (int i = 0; i < 7; ++i) { l[i] = logits[h * 7 + i]; m = fmaxf(m, l[i]); }
        float sum = 0.0f;
        for (int i = 0; i < 7; ++i) { l[i] = __expf(l[i] - m); sum += l[i]; }
        for (int i = 0; i < 7; ++i) l[i] = sqrtf(l[i] / sum);
        fw[h * FF + 0] = l[0];
        for (int n = 1; n <= 6; ++n) {
            fw[h * FF + 2 * n - 1] = l[n];
            fw[h * FF + 2 * n]     = l[n];
        }
    }
}

// ---------------------------------------------------------------------------
// Chebyshev mixture feature map: one wave32 per position, 2 dh per lane,
// wave shuffle reduction of the 13 feature sums.
// ---------------------------------------------------------------------------
__global__ __launch_bounds__(256) void psi_kernel(
    const __bf16* __restrict__ src, const float* __restrict__ fw,
    float* __restrict__ dst)
{
    const int lane = threadIdx.x & 31;
    const int wid  = blockIdx.x * (blockDim.x >> 5) + (threadIdx.x >> 5);
    const int nw   = gridDim.x * (blockDim.x >> 5);
    const int NPOS = BB * HH * SS;

    for (int p = wid; p < NPOS; p += nw) {
        const int h = (p >> 13) & (HH - 1);     // (p / S) % H
        const __bf16* xp = src + (size_t)p * DHH + lane * 2;
        float f[FF];
#pragma unroll
        for (int i = 0; i < FF; ++i) f[i] = 0.0f;
#pragma unroll
        for (int e = 0; e < 2; ++e) {
            float x  = (float)xp[e];
            float xc = fminf(1.0f, fmaxf(-1.0f, x));
            float sq = sqrtf(fmaxf(0.0f, 1.0f - xc * xc));
            float x2 = 2.0f * xc;
            float t0 = 1.0f, t1 = xc;
            float u0 = 1.0f, u1 = x2;
            f[0] += 1.0f;
            f[1] += t1;  f[2] += sq * u0;
            float t2 = x2 * t1 - t0;  f[3]  += t2;  f[4]  += sq * u1;
            float u2 = x2 * u1 - u0;
            float t3 = x2 * t2 - t1;  f[5]  += t3;  f[6]  += sq * u2;
            float u3 = x2 * u2 - u1;
            float t4 = x2 * t3 - t2;  f[7]  += t4;  f[8]  += sq * u3;
            float u4 = x2 * u3 - u2;
            float t5 = x2 * t4 - t3;  f[9]  += t5;  f[10] += sq * u4;
            float u5 = x2 * u4 - u3;
            float t6 = x2 * t5 - t4;  f[11] += t6;  f[12] += sq * u5;
        }
#pragma unroll
        for (int i = 0; i < FF; ++i)
#pragma unroll
            for (int off = 16; off >= 1; off >>= 1)
                f[i] += __shfl_xor(f[i], off, 32);
        if (lane == 0) {
            float* dp = dst + (size_t)p * FF;
            const float* fwp = fw + h * FF;
#pragma unroll
            for (int i = 0; i < FF; ++i) dp[i] = f[i] * fwp[i];
        }
    }
}

__global__ void zero_kernel(float* __restrict__ p, int n)
{
    int i = blockIdx.x * blockDim.x + threadIdx.x;
    if (i < n) p[i] = 0.0f;
}

// ---------------------------------------------------------------------------
// KV[b,h,f,d] = sum_s psiK[b,h,s,f] * v[b,h,s,d]   (chunked + f32 atomics)
// thread t: d = t&63, f-row tr = t>>6 covers f in {tr, tr+4, tr+8 (+12)}
// ---------------------------------------------------------------------------
__global__ __launch_bounds__(256) void kv_kernel(
    const float* __restrict__ psiK, const __bf16* __restrict__ vb,
    float* __restrict__ KV)
{
    const int t = threadIdx.x;
    const int d = t & 63, tr = t >> 6;
    const int CH = 8, SC = SS / CH;
    const int bh = blockIdx.x / CH, ch = blockIdx.x % CH;
    float a0 = 0.f, a1 = 0.f, a2 = 0.f, a3 = 0.f;
    const size_t base = (size_t)bh * SS;
    for (int s = ch * SC; s < (ch + 1) * SC; ++s) {
        float vv = (float)vb[(base + s) * DHH + d];
        const float* pp = psiK + (base + s) * FF;
        a0 += pp[tr + 0] * vv;
        a1 += pp[tr + 4] * vv;
        a2 += pp[tr + 8] * vv;
        if (tr == 0) a3 += pp[12] * vv;
    }
    float* kvp = KV + (size_t)bh * (FF * DHH);
    atomicAdd(&kvp[(tr + 0) * 64 + d], a0);
    atomicAdd(&kvp[(tr + 4) * 64 + d], a1);
    atomicAdd(&kvp[(tr + 8) * 64 + d], a2);
    if (tr == 0) atomicAdd(&kvp[12 * 64 + d], a3);
}

// ---------------------------------------------------------------------------
// ctx[b,s,h*64+d] = sum_f psiQ[b,h,s,f] * KV[b,h,f,d]  -> bf16 (B,S,D) layout
// ---------------------------------------------------------------------------
__global__ __launch_bounds__(256) void ctx_kernel(
    const float* __restrict__ psiQ, const float* __restrict__ KV,
    __bf16* __restrict__ ctxb)
{
    __shared__ float kvs[FF * DHH];
    const int t = threadIdx.x;
    const int CH = 32, SC = SS / CH;            // 256 positions per block
    const int bh = blockIdx.x / CH, ch = blockIdx.x % CH;
    const int b = bh >> 4, h = bh & 15;
    for (int i = t; i < FF * DHH; i += 256) kvs[i] = KV[(size_t)bh * (FF * DHH) + i];
    __syncthreads();
    const int d = t & 63, pr = t >> 6;          // 4 positions per iteration
    for (int i = 0; i < SC / 4; ++i) {
        int s = ch * SC + i * 4 + pr;
        const float* pp = psiQ + ((size_t)bh * SS + s) * FF;
        float c = 0.0f;
#pragma unroll
        for (int f = 0; f < FF; ++f) c += pp[f] * kvs[f * 64 + d];
        ctxb[((size_t)(b * SS + s)) * DD + h * DHH + d] = (__bf16)c;
    }
}

// ---------------------------------------------------------------------------
extern "C" void kernel_launch(void* const* d_in, const int* in_sizes, int n_in,
                              void* d_out, int out_size, void* d_ws, size_t ws_size,
                              hipStream_t stream)
{
    (void)in_sizes; (void)n_in; (void)out_size; (void)ws_size;
    const float* x  = (const float*)d_in[0];
    const float* Wq = (const float*)d_in[1];
    const float* bq = (const float*)d_in[2];
    const float* Wk = (const float*)d_in[3];
    const float* bk = (const float*)d_in[4];
    const float* Wv = (const float*)d_in[5];
    const float* bv = (const float*)d_in[6];
    const float* Wo = (const float*)d_in[7];
    const float* bo = (const float*)d_in[8];
    const float* dl = (const float*)d_in[9];
    float* out = (float*)d_out;

    char* ws = (char*)d_ws;
    size_t off = 0;
    auto take = [&](size_t bytes) -> char* {
        char* p = ws + off;
        off = (off + bytes + 255) & ~(size_t)255;
        return p;
    };
    __bf16* Xb   = (__bf16*)take((size_t)MTOT * DD * 2);        // 64 MB
    __bf16* Wt3  = (__bf16*)take((size_t)NQKV * DD * 2);        // 6 MB (N-major)
    __bf16* Wot  = (__bf16*)take((size_t)DD * DD * 2);          // 2 MB (N-major)
    float*  qkvb = (float*) take((size_t)NQKV * 4);
    float*  fw   = (float*) take((size_t)HH * FF * 4);
    __bf16* qn   = (__bf16*)take((size_t)BB * HH * SS * DHH * 2);
    __bf16* kn   = (__bf16*)take((size_t)BB * HH * SS * DHH * 2);
    __bf16* vbuf = (__bf16*)take((size_t)BB * HH * SS * DHH * 2);
    float*  psiQ = (float*) take((size_t)BB * HH * SS * FF * 4);
    float*  psiK = (float*) take((size_t)BB * HH * SS * FF * 4);
    float*  KV   = (float*) take((size_t)BB * HH * FF * DHH * 4);
    __bf16* ctxb = (__bf16*)take((size_t)MTOT * DD * 2);

    // 1) precision conversion + weight transposes
    cvt_x_kernel<<<(MTOT * DD / 4 + 255) / 256, 256, 0, stream>>>(x, Xb, MTOT * DD / 4);
    dim3 tb(32, 8);
    transpose_cvt_kernel<<<dim3(DD / 32, DD / 32), tb, 0, stream>>>(Wq, Wt3, DD, DD);
    transpose_cvt_kernel<<<dim3(DD / 32, DD / 32), tb, 0, stream>>>(Wk, Wt3 + (size_t)DD * DD, DD, DD);
    transpose_cvt_kernel<<<dim3(DD / 32, DD / 32), tb, 0, stream>>>(Wv, Wt3 + (size_t)2 * DD * DD, DD, DD);
    transpose_cvt_kernel<<<dim3(DD / 32, DD / 32), tb, 0, stream>>>(Wo, Wot, DD, DD);
    bias_kernel<<<(NQKV + 255) / 256, 256, 0, stream>>>(bq, bk, bv, qkvb);
    fw_kernel<<<1, 32, 0, stream>>>(dl, fw);

    // 2) fused QKV projection GEMM (bf16 WMMA, async LDS pipeline)
    const size_t smem = 128 * CT_PITCH * sizeof(float);   // 131.6 KB (>= 3 bufs)
    wmma_gemm_kernel<0><<<dim3(NQKV / 256, MTOT / 128), 256, smem, stream>>>(
        Xb, Wt3, DD, NQKV, qkvb, qn, kn, vbuf, nullptr);

    // 3) Chebyshev feature maps
    psi_kernel<<<4096, 256, 0, stream>>>(kn, fw, psiK);
    psi_kernel<<<4096, 256, 0, stream>>>(qn, fw, psiQ);

    // 4) KV = psiK^T v  (per b,h), then ctx = psiQ KV fused with relayout
    zero_kernel<<<(BB * HH * FF * DHH + 255) / 256, 256, 0, stream>>>(KV, BB * HH * FF * DHH);
    kv_kernel<<<BB * HH * 8, 256, 0, stream>>>(psiK, vbuf, KV);
    ctx_kernel<<<BB * HH * 32, 256, 0, stream>>>(psiQ, KV, ctxb);

    // 5) output projection GEMM (bf16 WMMA) + bo, f32 out
    wmma_gemm_kernel<1><<<dim3(DD / 256, MTOT / 128), 256, smem, stream>>>(
        ctxb, Wot, DD, DD, bo, nullptr, nullptr, nullptr, out);
}
